// TernaryLinear_63883343560960
// MI455X (gfx1250) — compile-verified, matrix operand order checked by reading, not run on hardware
//
#include <hip/hip_runtime.h>
#include <cstdint>

// Problem constants (from reference): B=4, M=2048, K=4096, N=4096
#define B_   4
#define M_   2048
#define K_   4096
#define N_   4096
#define R_   (B_ * M_)          // 8192 flattened rows of x / out

// GEMM tiling
#define TILE      128           // block tile: 128 (M) x 128 (N)
#define KT        64            // K-slab per LDS tile (2 WMMA k-steps of 32)
#define LDSK      72            // KT + 8 bf16 pad -> 144B row stride (16B aligned, de-aliased banks)
#define BUF_ELEMS (TILE * LDSK) // bf16 elements per matrix buffer
#define SMEM_BYTES (4 * BUF_ELEMS * 2) // 2 dbl-buffers x (A + B) x sizeof(bf16) = 73728 B

typedef __bf16 v16bf __attribute__((ext_vector_type(16)));
typedef __bf16 v8bf  __attribute__((ext_vector_type(8)));
typedef float  v8f   __attribute__((ext_vector_type(8)));

// ---------------------------------------------------------------------------
// fp32 -> bf16 round-to-nearest-even
// ---------------------------------------------------------------------------
__device__ __forceinline__ unsigned short f2bf(float f) {
    union { float f; unsigned u; } c; c.f = f;
    unsigned u = c.u;
    unsigned r = u + 0x7FFFu + ((u >> 16) & 1u);
    return (unsigned short)(r >> 16);
}

// Prepass 1: x (R_ x K_) fp32 -> bf16, same layout. One thread = 4 elements.
__global__ void cvt_x_bf16(const float* __restrict__ x,
                           unsigned long long* __restrict__ xb) {
    long i = (long)blockIdx.x * blockDim.x + threadIdx.x;   // i over R_*K_/4
    const float4 v = reinterpret_cast<const float4*>(x)[i];
    unsigned long long p =
          (unsigned long long)f2bf(v.x)
        | ((unsigned long long)f2bf(v.y) << 16)
        | ((unsigned long long)f2bf(v.z) << 32)
        | ((unsigned long long)f2bf(v.w) << 48);
    xb[i] = p;
}

// Prepass 2: W (K_ x N_) fp32 -> Wt (N_ x K_) bf16 (transpose so the GEMM's
// B-operand is K-contiguous like A). Writes coalesced along K; strided reads
// are a one-time 67MB cost.
__global__ void cvt_w_t_bf16(const float* __restrict__ w,
                             unsigned long long* __restrict__ wt) {
    long i = (long)blockIdx.x * blockDim.x + threadIdx.x;   // i over N_*K_/4
    const int K4 = K_ / 4;
    int n = (int)(i / K4);
    int k = (int)(i % K4) * 4;
    float a0 = w[(long)(k + 0) * N_ + n];
    float a1 = w[(long)(k + 1) * N_ + n];
    float a2 = w[(long)(k + 2) * N_ + n];
    float a3 = w[(long)(k + 3) * N_ + n];
    unsigned long long p =
          (unsigned long long)f2bf(a0)
        | ((unsigned long long)f2bf(a1) << 16)
        | ((unsigned long long)f2bf(a2) << 32)
        | ((unsigned long long)f2bf(a3) << 48);
    wt[i] = p;   // wt[n*K_ + k] (i == n*K4 + k/4)
}

// ---------------------------------------------------------------------------
// Async global->LDS tile fill: 128 rows x 64 bf16 (16KB), K-contiguous source.
// 256 threads x 4 x b128 (16B) each. Tracked by ASYNCcnt.
// ---------------------------------------------------------------------------
__device__ __forceinline__ void async_fill_tile(const __bf16* __restrict__ g,
                                                __bf16* s, int k0, int tid) {
#pragma unroll
    for (int i = 0; i < 4; ++i) {
        int id  = tid + i * 256;        // 0..1023 chunk id
        int row = id >> 3;              // 8 chunks of 8 bf16 per 64-wide row
        int kc  = (id & 7) << 3;
        unsigned long long ga =
            (unsigned long long)(uintptr_t)(const void*)(g + (long)row * K_ + k0 + kc);
        unsigned la =
            (unsigned)(unsigned long long)(uintptr_t)(void*)(s + row * LDSK + kc);
        asm volatile("global_load_async_to_lds_b128 %0, %1, off"
                     :: "v"(la), "v"(ga) : "memory");
    }
}

__device__ __forceinline__ void wait_async0() {
    asm volatile("s_wait_asynccnt 0" ::: "memory");
}

// 16 contiguous bf16 from LDS (two 16B-aligned halves)
__device__ __forceinline__ v16bf load_frag(const __bf16* p) {
    const v8bf* q = reinterpret_cast<const v8bf*>(p);
    v8bf lo = q[0];
    v8bf hi = q[1];
    return __builtin_shufflevector(lo, hi, 0, 1, 2, 3, 4, 5, 6, 7,
                                           8, 9, 10, 11, 12, 13, 14, 15);
}

// ---------------------------------------------------------------------------
// WMMA bf16 GEMM: out[r,n] = sum_k Xb[r,k] * Wt[n,k]
//   block = 128x128 tile, 256 threads = 8 waves (4 along M x 2 along N)
//   wave  = 32x64 region = 2x4 tiles of v_wmma_f32_16x16x32_bf16
//   LDS layout: [A buf0][B buf0][A buf1][B buf1], each BUF_ELEMS bf16
// ---------------------------------------------------------------------------
__global__ void __launch_bounds__(256)
ternary_wmma_gemm(const __bf16* __restrict__ Xb,
                  const __bf16* __restrict__ Wt,
                  float* __restrict__ out) {
    extern __shared__ __align__(16) __bf16 smem[];

    const int tid  = threadIdx.x;
    const int lane = tid & 31;
    const int wave = tid >> 5;
    const int wm   = wave & 3;      // wave's M strip (32 rows)
    const int wn   = wave >> 2;     // wave's N strip (64 cols)
    const int half = lane >> 4;     // K-halves / M-halves per ISA layouts
    const int lr   = lane & 15;

    const int cN = blockIdx.x * TILE;
    const int cM = blockIdx.y * TILE;

    const __bf16* gA = Xb + (long)cM * K_;
    const __bf16* gB = Wt + (long)cN * K_;

    v8f acc[2][4];
#pragma unroll
    for (int i = 0; i < 2; ++i)
#pragma unroll
        for (int j = 0; j < 4; ++j)
            acc[i][j] = (v8f){0.f, 0.f, 0.f, 0.f, 0.f, 0.f, 0.f, 0.f};

    // Prologue: fill buffer 0.
    async_fill_tile(gA, smem, 0, tid);
    async_fill_tile(gB, smem + BUF_ELEMS, 0, tid);
    wait_async0();
    __syncthreads();

    const int KSTEPS = K_ / KT;     // 64
    for (int t = 0; t < KSTEPS; ++t) {
        const int buf = t & 1;
        __bf16* sAc = smem + buf * (2 * BUF_ELEMS);             // current A
        __bf16* sBc = sAc + BUF_ELEMS;                          // current B

        if (t + 1 < KSTEPS) {       // overlap next DMA with this tile's WMMAs
            __bf16* sAn = smem + (buf ^ 1) * (2 * BUF_ELEMS);   // next A
            async_fill_tile(gA, sAn, (t + 1) * KT, tid);
            async_fill_tile(gB, sAn + BUF_ELEMS, (t + 1) * KT, tid);
        }

#pragma unroll
        for (int ks = 0; ks < 2; ++ks) {
            const int kb = ks * 32 + half * 16;   // lane's 16 contiguous K
            v16bf afrag[2], bfrag[4];
#pragma unroll
            for (int i = 0; i < 2; ++i)
                afrag[i] = load_frag(sAc + (wm * 32 + i * 16 + lr) * LDSK + kb);
#pragma unroll
            for (int j = 0; j < 4; ++j)
                bfrag[j] = load_frag(sBc + (wn * 64 + j * 16 + lr) * LDSK + kb);
#pragma unroll
            for (int i = 0; i < 2; ++i)
#pragma unroll
                for (int j = 0; j < 4; ++j)
                    acc[i][j] = __builtin_amdgcn_wmma_f32_16x16x32_bf16(
                        false, afrag[i], false, bfrag[j],
                        (short)0, acc[i][j], false, false);
        }

        wait_async0();              // next-tile DMA done before it is consumed
        __syncthreads();            // and before buf is overwritten next iter
    }

    // C/D layout (ISA 7.12.2): lanes 0-15 -> M=v, N=lane; lanes 16-31 -> M=8+v
#pragma unroll
    for (int i = 0; i < 2; ++i) {
#pragma unroll
        for (int j = 0; j < 4; ++j) {
            const int row0 = cM + wm * 32 + i * 16 + half * 8;
            const int col  = cN + wn * 64 + j * 16 + lr;
            float* o = out + (long)row0 * N_ + col;
#pragma unroll
            for (int v = 0; v < 8; ++v)
                o[(long)v * N_] = acc[i][j][v];
        }
    }
}

// ---------------------------------------------------------------------------
extern "C" void kernel_launch(void* const* d_in, const int* in_sizes, int n_in,
                              void* d_out, int out_size, void* d_ws, size_t ws_size,
                              hipStream_t stream) {
    (void)in_sizes; (void)n_in; (void)out_size; (void)ws_size;

    const float* x = (const float*)d_in[0];     // (B,M,K) fp32
    const float* w = (const float*)d_in[1];     // (K,N)  fp32 ternary
    float* out = (float*)d_out;                 // (B,M,N) fp32

    // Workspace: [ x_bf16 : R_*K_ ][ wT_bf16 : N_*K_ ]  (~100 MB, L2-resident)
    __bf16* xb = reinterpret_cast<__bf16*>(d_ws);
    __bf16* wt = xb + (size_t)R_ * K_;

    {   // x -> bf16
        const long n4 = (long)R_ * K_ / 4;          // 8,388,608
        cvt_x_bf16<<<(unsigned)(n4 / 256), 256, 0, stream>>>(
            x, reinterpret_cast<unsigned long long*>(xb));
    }
    {   // W -> W^T bf16
        const long n4 = (long)N_ * K_ / 4;          // 4,194,304
        cvt_w_t_bf16<<<(unsigned)(n4 / 256), 256, 0, stream>>>(
            w, reinterpret_cast<unsigned long long*>(wt));
    }

    dim3 grid(N_ / TILE, R_ / TILE);                // (32, 64)
    ternary_wmma_gemm<<<grid, 256, SMEM_BYTES, stream>>>(xb, wt, out);
}
